// MarketVQC_BERT_69028714381557
// MI455X (gfx1250) — matrix operand order repaired and checked by reading, not compiled, not run on hardware
//
#include <hip/hip_runtime.h>
#include <hip/hip_bf16.h>

// ---------------------------------------------------------------------------
// MarketVQC: proj(384->8)+tanh -> 8-qubit VQC -> <Z> -> decode(8->3)
// Restructured as: angles -> rank-1 product state psi (256) -> Phi = psi @ U^T
// (U = fixed 256x256 circuit unitary, built on device from q_weights) ->
// out_k = sum_z Phi_z^2 * c[k][z] + b_dec[k].
// The 65536x256x256 GEMM runs on v_wmma_f32_16x16x32_bf16 with B-fragment
// double-buffering so LDS fetches overlap matrix-core execution.
// ---------------------------------------------------------------------------

typedef __attribute__((ext_vector_type(16))) __bf16 v16bf;
typedef __attribute__((ext_vector_type(8)))  float  v8f;

#define NQ        8
#define NL        4
#define DIM       384
#define NSTATE    256
#define WG_ROWS   128
#define HALF_PI_F 1.570795f   // 3.14159 / 2, matches reference constant

union Frag16 { uint4 u[2]; v16bf v; };

// ---------------------------------------------------------------------------
// Kernel 1: build circuit unitary U (256x256) from q_weights, emit it as a
// bf16 blob in WMMA B-fragment order, plus the 3x256 parity-decode table c.
// One workgroup, 256 threads; thread t owns column t of U (column-local ops,
// so the evolution needs no barriers).
// ---------------------------------------------------------------------------
__global__ void vqc_build_unitary_kernel(const float* __restrict__ qw,
                                         const float* __restrict__ Wdec,
                                         __hip_bfloat16* __restrict__ ufrag,
                                         float* __restrict__ cz) {
  extern __shared__ float U[];            // 256*256 floats = 256 KB LDS
  const int t = threadIdx.x;              // column index

  // identity: column t evolves from basis state |t>
  for (int z = 0; z < NSTATE; ++z) U[z * NSTATE + t] = (z == t) ? 1.f : 0.f;

  // qubit q maps to bit position (7-q) of the flattened index (row-major (2,)*8)
  for (int l = 0; l < NL; ++l) {
    // per-qubit RY(q_weights[l][q])
    for (int q = 0; q < NQ; ++q) {
      const int p = 7 - q;
      const int m = 1 << p;
      const float th = qw[l * NQ + q] * 0.5f;
      const float c = cosf(th), s = sinf(th);
      for (int i = 0; i < NSTATE / 2; ++i) {
        const int z0 = ((i >> p) << (p + 1)) | (i & (m - 1));
        const int z1 = z0 | m;
        const float a0 = U[z0 * NSTATE + t];
        const float a1 = U[z1 * NSTATE + t];
        U[z0 * NSTATE + t] = c * a0 - s * a1;
        U[z1 * NSTATE + t] = s * a0 + c * a1;
      }
    }
    // ring CNOTs: control q, target (q+1)%8
    for (int q = 0; q < NQ; ++q) {
      const int tq = (q + 1) & 7;
      const int cb = 1 << (7 - q);
      const int tb = 1 << (7 - tq);
      for (int z = 0; z < NSTATE; ++z) {
        if ((z & cb) && !(z & tb)) {
          const float a = U[z * NSTATE + t];
          const float b = U[(z | tb) * NSTATE + t];
          U[z * NSTATE + t] = b;
          U[(z | tb) * NSTATE + t] = a;
        }
      }
    }
  }
  __syncthreads();   // blob phase reads other threads' columns

  // Emit bf16 blob in B-fragment order for v_wmma_f32_16x16x32_bf16:
  //   frag f = nt*8+ks covers N = nt*16..+15, K = ks*32..+31.
  //   lane L holds N = nt*16+(L&15); K-halves j=0..15 -> K = ks*32 + ((L&16)?16:0) + j.
  // Element value is UB[k][n] = U[n*256 + k]  (Phi_n = sum_k U[n][k] * psi_k).
  for (int i = 0; i < NSTATE; ++i) {
    const int e    = i * NSTATE + t;        // coalesced across t
    const int j    = e & 15;
    const int lane = (e >> 4) & 31;
    const int ks   = (e >> 9) & 7;
    const int nt   = e >> 12;
    const int n    = nt * 16 + (lane & 15);
    const int k    = ks * 32 + ((lane & 16) ? 16 : 0) + j;
    ufrag[e] = __float2bfloat16(U[n * NSTATE + k]);
  }

  // c[k][z] = sum_q W_dec[k][q] * (-1)^{bit(7-q) of z}
  for (int i = t; i < 3 * NSTATE; i += 256) {
    const int k = i >> 8;
    const int z = i & 255;
    float acc = 0.f;
    for (int q = 0; q < NQ; ++q)
      acc += Wdec[k * NQ + q] * (((z >> (7 - q)) & 1) ? -1.f : 1.f);
    cz[i] = acc;
  }
}

// ---------------------------------------------------------------------------
// Kernel 2: fused projection + product-state build + WMMA GEMM + decode.
// 256 threads (8 waves) per WG, 128 batch rows per WG.
// ---------------------------------------------------------------------------
__global__ void vqc_main_kernel(const float* __restrict__ x,
                                const float* __restrict__ Wp,
                                const float* __restrict__ bp,
                                const __hip_bfloat16* __restrict__ ufrag,
                                const float* __restrict__ cz,
                                const float* __restrict__ bdec,
                                float* __restrict__ out) {
  extern __shared__ char smem[];
  float*          wp   = (float*)(smem);                    // 8*384 f32   = 12288 B
  float*          part = (float*)(smem + 12288);            // 256*8 f32   =  8192 B
  float*          cs   = (float*)(smem + 20480);            // 128*16 f32  =  8192 B
  __hip_bfloat16* psi  = (__hip_bfloat16*)(smem + 28672);   // 128*256 bf16= 65536 B
  uint4*          ubl4 = (uint4*)(smem + 94208);            // U blob      =131072 B
  float*          czl  = (float*)(smem + 225280);           // 3*256 f32   =  3072 B
  float*          oacc = (float*)(smem + 228352);           // 128*3 f32   =  1536 B
                                                            // total 229888 B

  const int  t       = threadIdx.x;
  const long rowBase = (long)blockIdx.x * WG_ROWS;

  // ---- stage constants into LDS (coalesced) ----
  {
    const uint4* usrc = (const uint4*)ufrag;                 // 8192 uint4
    for (int i = t; i < 8192; i += 256) ubl4[i] = usrc[i];
    for (int i = t; i < NQ * DIM; i += 256) wp[i] = Wp[i];
    for (int i = t; i < 3 * NSTATE; i += 256) czl[i] = cz[i];
    if (t < 3 * WG_ROWS) oacc[t] = 0.f;
  }
  __syncthreads();

  // ---- phase 1: projection partials (2 threads per row, split K) ----
  {
    const int row = t & 127;
    const int kh  = t >> 7;                                   // K half: 0 or 1
    const float*  xr  = x + (rowBase + row) * DIM + kh * (DIM / 2);
    const float4* xv4 = (const float4*)xr;
    __builtin_prefetch(xr + 96, 0, 0);                        // global_prefetch_b8
    float acc[8];
#pragma unroll
    for (int q = 0; q < 8; ++q) acc[q] = 0.f;
    for (int kk = 0; kk < DIM / 8; ++kk) {                    // 48 float4 chunks
      const float4 xv = xv4[kk];
      const int k = kh * (DIM / 2) + kk * 4;
#pragma unroll
      for (int q = 0; q < 8; ++q) {
        const float* w = wp + q * DIM + k;
        acc[q] += xv.x * w[0] + xv.y * w[1] + xv.z * w[2] + xv.w * w[3];
      }
    }
#pragma unroll
    for (int q = 0; q < 8; ++q) part[t * 8 + q] = acc[q];
  }
  __syncthreads();

  // ---- phase 1b: tanh, angles, cos/sin of half-angles ----
  if (t < WG_ROWS) {
#pragma unroll
    for (int q = 0; q < 8; ++q) {
      const float s = part[t * 8 + q] + part[(t + 128) * 8 + q] + bp[q];
      const float h = tanhf(s);
      const float a = (h + 1.f) * (HALF_PI_F * 0.5f);         // half of angle
      cs[t * 16 + q * 2 + 0] = cosf(a);
      cs[t * 16 + q * 2 + 1] = sinf(a);
    }
  }
  __syncthreads();

  // ---- phase 1c: product state psi[row][z] (bf16), z-bit b <-> qubit (7-b) ----
  {
    const int row  = t & 127;
    const int half = t >> 7;                                  // z bit 7 (= qubit 0)
    const float* mycs = cs + row * 16;
    float lo[16], hi[8];
    lo[0] = 1.f;
    int size = 1;
#pragma unroll
    for (int j = 0; j < 4; ++j) {                             // z bits 0..3 -> qubits 7..4
      const float c = mycs[(7 - j) * 2 + 0];
      const float s = mycs[(7 - j) * 2 + 1];
      for (int m = size - 1; m >= 0; --m) { lo[m + size] = lo[m] * s; lo[m] = lo[m] * c; }
      size <<= 1;
    }
    hi[0] = mycs[0 * 2 + half];                               // qubit 0 factor
    size = 1;
#pragma unroll
    for (int j = 0; j < 3; ++j) {                             // z bits 4..6 -> qubits 3..1
      const float c = mycs[(3 - j) * 2 + 0];
      const float s = mycs[(3 - j) * 2 + 1];
      for (int m = size - 1; m >= 0; --m) { hi[m + size] = hi[m] * s; hi[m] = hi[m] * c; }
      size <<= 1;
    }
    __hip_bfloat16* prow = psi + row * NSTATE + half * 128;
    for (int i = 0; i < 128; ++i)
      prow[i] = __float2bfloat16(lo[i & 15] * hi[i >> 4]);
  }
  __syncthreads();

  // ---- phase 2: Phi = Psi @ U^T via WMMA; square; fold decode table ----
  {
    const int wave  = t >> 5;
    const int lane  = t & 31;
    const int rbase = wave * 16;                              // 16 batch rows / wave
    const int arow  = rbase + (lane & 15);
    const int abit  = (lane & 16) ? 8 : 0;                    // A-frag K sub-block
    const uint4* psiRow = (const uint4*)(psi + arow * NSTATE);

    float facc[8][3];
#pragma unroll
    for (int v = 0; v < 8; ++v) { facc[v][0] = 0.f; facc[v][1] = 0.f; facc[v][2] = 0.f; }

    for (int nt = 0; nt < 16; ++nt) {                         // 16 N-tiles of 16
      v8f acc = {};
      // B fragments for this N-tile: 8 K-steps, 64 uint4 apart per lane.
      const uint4* bbase = ubl4 + ((nt * 8) * 32 + lane) * 2;
      Frag16 Bcur;
      Bcur.u[0] = bbase[0];
      Bcur.u[1] = bbase[1];
#pragma unroll
      for (int ks = 0; ks < 8; ++ks) {                        // K = 256 in 8 steps
        // double-buffer: issue next B loads before the WMMA consuming Bcur,
        // so ds fetch latency hides under matrix-core execution.
        Frag16 Bnext = Bcur;
        if (ks < 7) {
          Bnext.u[0] = bbase[(ks + 1) * 64 + 0];
          Bnext.u[1] = bbase[(ks + 1) * 64 + 1];
        }
        // A fragment: lane<16 -> K = ks*32 + {0..15}; lane>=16 -> +16
        const int base0 = ks * 32 + 2 * abit;                 // halves
        Frag16 A;
        A.u[0] = psiRow[(base0 >> 3) + 0];
        A.u[1] = psiRow[(base0 >> 3) + 1];
        acc = __builtin_amdgcn_wmma_f32_16x16x32_bf16(
            false, A.v, false, Bcur.v, (short)0, acc, false, false);
        Bcur = Bnext;
      }
      // D layout: lane holds column n = nt*16+(lane&15); VGPR v -> row v (+8 hi lanes)
      const int z  = nt * 16 + (lane & 15);
      const float c0 = czl[0 * NSTATE + z];
      const float c1 = czl[1 * NSTATE + z];
      const float c2 = czl[2 * NSTATE + z];
#pragma unroll
      for (int v = 0; v < 8; ++v) {
        const float p = acc[v] * acc[v];
        facc[v][0] += p * c0;
        facc[v][1] += p * c1;
        facc[v][2] += p * c2;
      }
    }
    // cross-lane reduce (16 lanes share each row) via LDS f32 atomics
#pragma unroll
    for (int v = 0; v < 8; ++v) {
      const int r = rbase + v + abit;
      atomicAdd(&oacc[r * 3 + 0], facc[v][0]);
      atomicAdd(&oacc[r * 3 + 1], facc[v][1]);
      atomicAdd(&oacc[r * 3 + 2], facc[v][2]);
    }
  }
  __syncthreads();

  // ---- phase 3: out = feats @ W_dec^T + b_dec (already folded; add bias) ----
  if (t < 3 * WG_ROWS) {
    const int k = t % 3;
    out[rowBase * 3 + t] = oacc[t] + bdec[k];
  }
}

// ---------------------------------------------------------------------------
extern "C" void kernel_launch(void* const* d_in, const int* in_sizes, int n_in,
                              void* d_out, int out_size, void* d_ws, size_t ws_size,
                              hipStream_t stream) {
  const float* x   = (const float*)d_in[0];   // (B, 384)
  const float* Wp  = (const float*)d_in[1];   // (8, 384)
  const float* bp  = (const float*)d_in[2];   // (8,)
  const float* qw  = (const float*)d_in[3];   // (4, 8)
  const float* Wd  = (const float*)d_in[4];   // (3, 8)
  const float* bd  = (const float*)d_in[5];   // (3,)
  float* out = (float*)d_out;

  __hip_bfloat16* ufrag = (__hip_bfloat16*)d_ws;                   // 256*256 bf16
  float*          cz    = (float*)((char*)d_ws + NSTATE * NSTATE * 2); // 3*256 f32

  const int B      = in_sizes[0] / DIM;
  const int blocks = B / WG_ROWS;

  vqc_build_unitary_kernel<<<1, 256, NSTATE * NSTATE * sizeof(float), stream>>>(
      qw, Wd, ufrag, cz);
  vqc_main_kernel<<<blocks, 256, 229888, stream>>>(
      x, Wp, bp, ufrag, cz, bd, out);
}
